// LlamaAttention_heavy_hitter_15358803051032
// MI455X (gfx1250) — matrix operand order repaired and checked by reading, not compiled, not run on hardware
//
#include <hip/hip_runtime.h>
#include <hip/hip_bf16.h>
#include <stdint.h>

// ---------------------------------------------------------------------------
// LlamaAttention + H2O heavy-hitter mask for MI455X (gfx1250, wave32, WMMA)
//
// Shapes: B=1, S=1024, NH=32, HD=128, HIDDEN=4096
// heavy_budget = recent_budget = 102, cache_budget = 204, penalty = 0.99
//
// MI455X reasoning: projection GEMMs are bandwidth bound; bf16 halves bytes
// and v_wmma_f32_16x16x32_bf16 (f32 accumulate) gives far more math than
// needed. GEMM uses 128x128 block tiles; k-slices are staged into
// double-buffered LDS by the Tensor Data Mover (tensor_load_to_lds, TENSORcnt)
// when available -- the DMA runs while WMMAs consume the other buffer -- with
// a register-staging fallback. The D# pad feature (16B pad per 64B) gives the
// 80B LDS row stride used for conflict-free ds_load_b128. The serial H2O scan
// is latency-bound: one 1024-thread block per head, select state in
// registers, top-k via binary search on float bit ordering.
// ---------------------------------------------------------------------------

#define S_LEN    1024
#define NHEADS   32
#define HDIM     128
#define HIDDEN_  (NHEADS * HDIM)   // 4096
#define HEAVY_B  102
#define RECENT_B 102
#define CACHE_B  (HEAVY_B + RECENT_B)
#define PENALTY_F 0.99f
#define NEG_MASK (-1.0e9f)

typedef __attribute__((ext_vector_type(16))) __bf16 v16bf;
typedef __attribute__((ext_vector_type(8)))  float  v8f;

union FragU { v16bf v; uint4 q[2]; };

#if defined(__HIP_DEVICE_COMPILE__) && __has_builtin(__builtin_amdgcn_tensor_load_to_lds) && __has_builtin(__builtin_amdgcn_s_wait_tensorcnt)
#define USE_TDM 1
#else
#define USE_TDM 0
#endif

#if USE_TDM
typedef __attribute__((ext_vector_type(4))) unsigned int u32x4_t;
typedef __attribute__((ext_vector_type(8))) int          i32x8_t;
typedef __attribute__((ext_vector_type(4))) int          i32x4_t;

// Issue one TDM 2D tile load: tile_h rows x tile_w elements (2-byte data),
// row stride row_stride_elems, destination LDS byte address lds_byte_addr.
// Padding: 4 DWORDs (16B) inserted after every 16 DWORDs (64B) stored, which
// turns a 64B data row into the 80B LDS row stride of As/Bs below.
// D# layout per CDNA5 ISA ch.8 (group0 128b, group1 256b; remaining groups
// unused for 2D tiles and passed zero-filled to the 6-arg builtin).
__device__ __forceinline__ void tdm_load_tile_2d(const unsigned short* gsrc,
                                                 unsigned lds_byte_addr,
                                                 unsigned tile_w_elems,
                                                 unsigned tile_h_rows,
                                                 unsigned row_stride_elems)
{
    unsigned long long ga = (unsigned long long)(uintptr_t)gsrc;
    u32x4_t g0;
    g0[0] = 1u;                                   // count=1 (valid descriptor)
    g0[1] = lds_byte_addr;                        // lds_addr
    g0[2] = (unsigned)(ga & 0xFFFFFFFFu);         // global_addr[31:0]
    g0[3] = (unsigned)((ga >> 32) & 0x01FFFFFFu)  // global_addr[56:32]
          | (2u << 30);                           // type=2 ("image")

    i32x8_t g1;
    g1[0] = (int)((1u << 16)                      // data_size=1 -> 2 bytes
                | (1u << 20)                      // pad_enable
                | (3u << 22)                      // pad_interval: 16 DWORDs (64B)
                | (3u << 25));                    // pad_amount: 4 DWORDs (16B)
    g1[1] = (int)((tile_w_elems & 0xFFFFu) << 16);                 // tensor_dim0 lo
    g1[2] = (int)(((tile_w_elems >> 16) & 0xFFFFu)
                | ((tile_h_rows & 0xFFFFu) << 16));                // dim0 hi | dim1 lo
    g1[3] = (int)(((tile_h_rows >> 16) & 0xFFFFu)
                | ((tile_w_elems & 0xFFFFu) << 16));               // dim1 hi | tile_dim0
    g1[4] = (int)(tile_h_rows & 0xFFFFu);                          // tile_dim1 (tile_dim2=0)
    g1[5] = (int)row_stride_elems;                                 // tensor_dim0_stride lo
    g1[6] = 0;                                                     // stride hi | dim1_stride
    g1[7] = 0;
    i32x4_t gz4 = (i32x4_t)0;
    i32x8_t gz8 = (i32x8_t)0;
    __builtin_amdgcn_tensor_load_to_lds(g0, g1, gz4, gz4, gz8, 0);
}
#endif

static __device__ __forceinline__ unsigned short f2bf(float f) {
    union { float f; unsigned int u; } c; c.f = f;
    unsigned int u = c.u;
    unsigned int r = u + 0x7FFFu + ((u >> 16) & 1u);   // round-to-nearest-even
    return (unsigned short)(r >> 16);
}

// ----------------------------- reductions ----------------------------------
__device__ __forceinline__ float wredMax(float v) {
#pragma unroll
    for (int o = 16; o > 0; o >>= 1) v = fmaxf(v, __shfl_down(v, o, 32));
    return v;
}
__device__ __forceinline__ float wredSum(float v) {
#pragma unroll
    for (int o = 16; o > 0; o >>= 1) v += __shfl_down(v, o, 32);
    return v;
}
__device__ __forceinline__ int wredSumI(int v) {
#pragma unroll
    for (int o = 16; o > 0; o >>= 1) v += __shfl_down(v, o, 32);
    return v;
}

__device__ __forceinline__ float blockMaxAll(float v, float* red) {
    const int w = threadIdx.x >> 5, l = threadIdx.x & 31, nw = blockDim.x >> 5;
    v = wredMax(v);
    if (l == 0) red[w] = v;
    __syncthreads();
    if (w == 0) {
        float t = (l < nw) ? red[l] : -3.0e38f;
        t = wredMax(t);
        if (l == 0) red[0] = t;
    }
    __syncthreads();
    float r = red[0];
    __syncthreads();
    return r;
}
__device__ __forceinline__ float blockSumAll(float v, float* red) {
    const int w = threadIdx.x >> 5, l = threadIdx.x & 31, nw = blockDim.x >> 5;
    v = wredSum(v);
    if (l == 0) red[w] = v;
    __syncthreads();
    if (w == 0) {
        float t = (l < nw) ? red[l] : 0.0f;
        t = wredSum(t);
        if (l == 0) red[0] = t;
    }
    __syncthreads();
    float r = red[0];
    __syncthreads();
    return r;
}
__device__ __forceinline__ int blockSumAllI(int v, int* red) {
    const int w = threadIdx.x >> 5, l = threadIdx.x & 31, nw = blockDim.x >> 5;
    v = wredSumI(v);
    if (l == 0) red[w] = v;
    __syncthreads();
    if (w == 0) {
        int t = (l < nw) ? red[l] : 0;
        t = wredSumI(t);
        if (l == 0) red[0] = t;
    }
    __syncthreads();
    int r = red[0];
    __syncthreads();
    return r;
}

// ----------------------- fp32 -> bf16 conversion ---------------------------
__global__ void cvt_f32_bf16(const float* __restrict__ in,
                             unsigned short* __restrict__ out, long long n) {
    long long i = (long long)blockIdx.x * blockDim.x + threadIdx.x;
    if (i < n) out[i] = f2bf(in[i]);
}

// --------------------------- WMMA GEMM -------------------------------------
// C[M,N] = scale * (A[M,K] @ Bt[N,K]^T), bf16 inputs, f32 accumulate.
// 256-thread block computes a 128x128 tile: 8 waves in a 4(M)x2(N) grid,
// each wave owns a 32x64 strip (2x4 WMMA accumulators). K is consumed in
// slices of 32 staged through double-buffered LDS (rows padded to 40 elems =
// 80B stride: 16B-aligned, bank-conflict-free b128 LDS accesses). The k+1
// slice transfer (TDM or global loads) is issued before the WMMAs of slice k.
// flags bit0: causal (col>row -> NEG_MASK);  bit1: output bf16 (else f32).
// Requires M%128==0, N%128==0, K%32==0 (all call sites satisfy this).

__device__ __forceinline__ void wmma_slice(const unsigned short (*Asb)[40],
                                           const unsigned short (*Bsb)[40],
                                           int wm, int wn, int amr, int abase,
                                           int bn, int bko, v8f (&acc)[2][4])
{
    FragU afr0, afr1;
    const unsigned short* ap0 = &Asb[(wm << 5) + amr][abase];
    afr0.q[0] = *(const uint4*)(ap0);
    afr0.q[1] = *(const uint4*)(ap0 + 16);
    const unsigned short* ap1 = &Asb[(wm << 5) + 16 + amr][abase];
    afr1.q[0] = *(const uint4*)(ap1);
    afr1.q[1] = *(const uint4*)(ap1 + 16);
#pragma unroll
    for (int j = 0; j < 4; ++j) {
        FragU b;
        const unsigned short* bp = &Bsb[(wn << 6) + (j << 4) + bn][bko];
        b.q[0] = *(const uint4*)(bp);
        b.q[1] = *(const uint4*)(bp + 16);
        acc[0][j] = __builtin_amdgcn_wmma_f32_16x16x32_bf16(
            false, afr0.v, false, b.v, (short)0, acc[0][j], false, false);
        acc[1][j] = __builtin_amdgcn_wmma_f32_16x16x32_bf16(
            false, afr1.v, false, b.v, (short)0, acc[1][j], false, false);
    }
}

__global__ void __launch_bounds__(256)
gemm_bf16_wmma(const unsigned short* __restrict__ A, int lda, long long sAz,
               const unsigned short* __restrict__ Bt, int ldb, long long sBz,
               void* __restrict__ Cv, int ldc, long long sCz,
               int M, int N, int K, float scale, int flags)
{
    __shared__ unsigned short As[2][128][40];   // [buf][row][k] (+8 pad = 80B stride)
    __shared__ unsigned short Bs[2][128][40];   // [buf][col][k] (+8 pad)

    const int tid  = threadIdx.x;
    const int lane = tid & 31;
    const int wid  = tid >> 5;          // 0..7
    const int wm   = wid >> 1;          // 0..3 : 32-row strip
    const int wn   = wid & 1;           // 0..1 : 64-col strip
    const int z    = blockIdx.z;
    const int m0b  = blockIdx.y << 7;
    const int n0b  = blockIdx.x << 7;

    const unsigned short* Ap = A + (long long)z * sAz;
    const unsigned short* Bp = Bt + (long long)z * sBz;

    // A 16x32 bf16 fragment: lanes 0-15 row m, K {0..7,16..23}; lanes 16-31 K {8..15,24..31}
    const int amr   = lane & 15;
    const int abase = (lane >> 4) << 3; // 0 or 8
    // B 32x16 bf16 fragment: lane&15 = column; lanes 0-15 K {0..7,16..23}; lanes 16-31 K {8..15,24..31}
    const int bn  = lane & 15;
    const int bko = (lane >> 4) << 3;   // 0 or 8

    v8f acc[2][4] = {};
    const int nk = K >> 5;
    int cur = 0;

#if USE_TDM
    // ---- Tensor Data Mover staging (TENSORcnt), wave 0 issues the DMAs ----
    if (wid == 0) {
        tdm_load_tile_2d(Ap + (long long)m0b * lda,
                         (unsigned)(uintptr_t)&As[0][0][0], 32u, 128u, (unsigned)lda);
        tdm_load_tile_2d(Bp + (long long)n0b * ldb,
                         (unsigned)(uintptr_t)&Bs[0][0][0], 32u, 128u, (unsigned)ldb);
        __builtin_amdgcn_s_wait_tensorcnt(0);
    }
    __syncthreads();
#pragma unroll 1
    for (int kt = 0; kt < nk; ++kt) {
        const bool more = (kt + 1 < nk);
        if (more && wid == 0) {            // DMA slice kt+1 while WMMAs run on kt
            const int k0n = (kt + 1) << 5;
            tdm_load_tile_2d(Ap + (long long)m0b * lda + k0n,
                             (unsigned)(uintptr_t)&As[cur ^ 1][0][0], 32u, 128u, (unsigned)lda);
            tdm_load_tile_2d(Bp + (long long)n0b * ldb + k0n,
                             (unsigned)(uintptr_t)&Bs[cur ^ 1][0][0], 32u, 128u, (unsigned)ldb);
        }
        if (kt + 2 < nk) {                 // gfx1250 global_prefetch for k+2 into L2
            const int kp = (kt + 2) << 5;
            __builtin_prefetch(Bp + (long long)(n0b + (tid >> 2)) * ldb + kp, 0, 1);
        }
        wmma_slice(As[cur], Bs[cur], wm, wn, amr, abase, bn, bko, acc);
        if (more) {
            if (wid == 0) __builtin_amdgcn_s_wait_tensorcnt(0);
            __syncthreads();
            cur ^= 1;
        }
    }
#else
    // ---- fallback: register staging (global_load_b128 -> ds_store_b128) ----
    const int ar = tid >> 2;            // 0..63
    const int aq = (tid & 3) << 3;      // 0,8,16,24 (element offset)
    uint4 ra0, ra1, rb0, rb1;
    ra0 = *(const uint4*)(Ap + (long long)(m0b + ar)      * lda + aq);
    ra1 = *(const uint4*)(Ap + (long long)(m0b + ar + 64) * lda + aq);
    rb0 = *(const uint4*)(Bp + (long long)(n0b + ar)      * ldb + aq);
    rb1 = *(const uint4*)(Bp + (long long)(n0b + ar + 64) * ldb + aq);
    *(uint4*)&As[0][ar][aq]      = ra0;
    *(uint4*)&As[0][ar + 64][aq] = ra1;
    *(uint4*)&Bs[0][ar][aq]      = rb0;
    *(uint4*)&Bs[0][ar + 64][aq] = rb1;
#pragma unroll 1
    for (int kt = 0; kt < nk; ++kt) {
        __syncthreads();
        const bool more = (kt + 1 < nk);
        if (more) {
            const int k0 = (kt + 1) << 5;
            ra0 = *(const uint4*)(Ap + (long long)(m0b + ar)      * lda + k0 + aq);
            ra1 = *(const uint4*)(Ap + (long long)(m0b + ar + 64) * lda + k0 + aq);
            rb0 = *(const uint4*)(Bp + (long long)(n0b + ar)      * ldb + k0 + aq);
            rb1 = *(const uint4*)(Bp + (long long)(n0b + ar + 64) * ldb + k0 + aq);
        }
        if (kt + 2 < nk) {
            const int kp = (kt + 2) << 5;
            __builtin_prefetch(Bp + (long long)(n0b + ar) * ldb + kp + aq, 0, 1);
        }
        wmma_slice(As[cur], Bs[cur], wm, wn, amr, abase, bn, bko, acc);
        if (more) {
            const int nxt = cur ^ 1;
            *(uint4*)&As[nxt][ar][aq]      = ra0;
            *(uint4*)&As[nxt][ar + 64][aq] = ra1;
            *(uint4*)&Bs[nxt][ar][aq]      = rb0;
            *(uint4*)&Bs[nxt][ar + 64][aq] = rb1;
            cur = nxt;
        }
    }
#endif

    // C 16x16 f32 layout: VGPR r, lane l -> (M = r + 8*(l>=16), N = l&15)
    const int rb_  = m0b + (wm << 5);
    const int cl   = lane & 15;
    const int rsub = (lane >> 4) << 3;
#pragma unroll
    for (int ms = 0; ms < 2; ++ms) {
#pragma unroll
        for (int j = 0; j < 4; ++j) {
#pragma unroll
            for (int r = 0; r < 8; ++r) {
                int rr = rb_ + (ms << 4) + rsub + r;
                int cc = n0b + (wn << 6) + (j << 4) + cl;
                float v = acc[ms][j][r] * scale;
                if ((flags & 1) && (cc > rr)) v = NEG_MASK;
                long long idx = (long long)z * sCz + (long long)rr * ldc + cc;
                if (flags & 2) ((unsigned short*)Cv)[idx] = f2bf(v);
                else           ((float*)Cv)[idx] = v;
            }
        }
    }
}

// ------------------------ RoPE + bf16 repack -------------------------------
// qf/kf: [S, HIDDEN] f32 -> qh/kh: [NH, S, HD] bf16 with RoPE applied.
__global__ void rope_cvt(const float* __restrict__ qf, const float* __restrict__ kf,
                         unsigned short* __restrict__ qh, unsigned short* __restrict__ kh)
{
    int idx = blockIdx.x * blockDim.x + threadIdx.x;    // NH * S * 64
    if (idx >= NHEADS * S_LEN * 64) return;
    int d = idx & 63;
    int s = (idx >> 6) & (S_LEN - 1);
    int h = idx >> 16;

    float inv = __expf(-(float)d * (9.210340371976184f / 64.0f)); // 10000^(-d/64)
    float ang = (float)s * inv;
    float sn, c;
    sincosf(ang, &sn, &c);

    long long gi = (long long)s * HIDDEN_ + h * HDIM + d;
    float q0 = qf[gi], q1 = qf[gi + 64];
    float k0 = kf[gi], k1 = kf[gi + 64];
    long long oi = ((long long)h * S_LEN + s) * HDIM + d;
    qh[oi]      = f2bf(q0 * c - q1 * sn);
    qh[oi + 64] = f2bf(q1 * c + q0 * sn);
    kh[oi]      = f2bf(k0 * c - k1 * sn);
    kh[oi + 64] = f2bf(k1 * c + k0 * sn);
}

// -------------------- V transpose to [NH, HD, S] bf16 ----------------------
__global__ void vT_cvt(const float* __restrict__ vf, unsigned short* __restrict__ vt)
{
    int idx = blockIdx.x * blockDim.x + threadIdx.x;    // NH * HD * S
    if (idx >= NHEADS * HDIM * S_LEN) return;
    int s = idx & (S_LEN - 1);
    int d = (idx >> 10) & (HDIM - 1);
    int h = idx >> 17;
    vt[idx] = f2bf(vf[(long long)s * HIDDEN_ + h * HDIM + d]);
}

// ------------------------- H2O heavy-hitter scan ---------------------------
// One block per head, 1024 threads (thread j owns key position j). Applies
// the running mask in place to attn rows; maintains `select` in registers;
// top-k via binary search over float bit-ordering (select >= 0 always).
__global__ void __launch_bounds__(1024)
hh_scan(float* __restrict__ attn)
{
    const int h = blockIdx.x;
    const int j = threadIdx.x;
    float* base = attn + (long long)h * S_LEN * S_LEN;
    __shared__ float redf[32];
    __shared__ int   redi[32];

    float sel = 0.0f;
    bool  m   = (j == 0);            // mask row 0: pos == 0

    for (int t = 0; t < S_LEN; ++t) {
        float a = base[(long long)t * S_LEN + j];
        base[(long long)t * S_LEN + j] = m ? a : NEG_MASK;   // final mask, in place
        if (t == S_LEN - 1) break;

        // softmax of where(cur_mask, row_t, -inf)
        float x  = m ? a : -3.0e38f;
        float mx = blockMaxAll(x, redf);
        float e  = m ? __expf(x - mx) : 0.0f;
        float sum = blockSumAll(e, redf);
        float tmp = e / sum;

        sel = PENALTY_F * sel + tmp;   // penalty accumulation
        if (!m) sel = 0.0f;            // select *= cur_mask

        const int local = t - RECENT_B;
        bool nm;
        if (t >= CACHE_B) {
            const bool valid = (j <= local);
            const unsigned key = valid ? __float_as_uint(sel) : 0u;
            // largest threshold T with count(valid && key >= T) >= HEAVY_B
            unsigned lo = 0u, hi = 0x7F800000u;
#pragma unroll 1
            for (int it = 0; it < 31; ++it) {
                if (lo >= hi) break;   // uniform across block
                unsigned mid = lo + ((hi - lo + 1u) >> 1);
                int cnt = blockSumAllI((valid && key >= mid) ? 1 : 0, redi);
                if (cnt >= HEAVY_B) lo = mid; else hi = mid - 1u;
            }
            bool heavy  = valid && (key >= lo);
            bool recent = (j > local) && (j <= t + 1);
            nm = heavy || recent;
        } else {
            nm = (j <= t + 1);         // causal row
        }
        m = nm;
    }
}

// ----------------------- row softmax -> bf16 probs -------------------------
__global__ void __launch_bounds__(256)
softmax_rows(const float* __restrict__ attn, unsigned short* __restrict__ P)
{
    __shared__ float redf[32];
    long long row = blockIdx.x;                 // h*S + t
    const float* a = attn + row * S_LEN;
    unsigned short* p = P + row * S_LEN;

    float vals[4];
    float vmax = -3.0e38f;
#pragma unroll
    for (int i = 0; i < 4; ++i) {
        vals[i] = a[threadIdx.x + (i << 8)];
        vmax = fmaxf(vmax, vals[i]);
    }
    vmax = blockMaxAll(vmax, redf);
    float s = 0.0f;
#pragma unroll
    for (int i = 0; i < 4; ++i) {
        vals[i] = __expf(vals[i] - vmax);
        s += vals[i];
    }
    s = blockSumAll(s, redf);
    float inv = 1.0f / s;
#pragma unroll
    for (int i = 0; i < 4; ++i)
        p[threadIdx.x + (i << 8)] = f2bf(vals[i] * inv);
}

// ---------------------------------------------------------------------------
extern "C" void kernel_launch(void* const* d_in, const int* in_sizes, int n_in,
                              void* d_out, int out_size, void* d_ws, size_t ws_size,
                              hipStream_t stream)
{
    (void)in_sizes; (void)n_in; (void)out_size; (void)ws_size;
    const float* hs = (const float*)d_in[0];   // [1,S,HIDDEN]
    // d_in[1] attention_mask, d_in[2] position_ids: folded analytically
    const float* Wq = (const float*)d_in[3];
    const float* Wk = (const float*)d_in[4];
    const float* Wv = (const float*)d_in[5];
    const float* Wo = (const float*)d_in[6];

    char* ws = (char*)d_ws;
    size_t off = 0;
    auto alloc = [&](size_t bytes) -> char* {
        char* p = ws + off;
        off += (bytes + 255) & ~(size_t)255;
        return p;
    };

    const long long nAct = (long long)S_LEN * HIDDEN_;       // 4,194,304
    const long long nW   = (long long)HIDDEN_ * HIDDEN_;     // 16,777,216

    unsigned short* hsb = (unsigned short*)alloc(nAct * 2);
    unsigned short* wqb = (unsigned short*)alloc(nW * 2);
    unsigned short* wkb = (unsigned short*)alloc(nW * 2);
    unsigned short* wvb = (unsigned short*)alloc(nW * 2);
    unsigned short* wob = (unsigned short*)alloc(nW * 2);
    float* qf = (float*)alloc(nAct * 4);
    float* kf = (float*)alloc(nAct * 4);
    float* vf = (float*)alloc(nAct * 4);
    unsigned short* qh = (unsigned short*)alloc(nAct * 2);   // [NH,S,HD]
    unsigned short* kh = (unsigned short*)alloc(nAct * 2);
    unsigned short* vt = (unsigned short*)alloc(nAct * 2);   // [NH,HD,S]
    float* attn = (float*)alloc((long long)NHEADS * S_LEN * S_LEN * 4);
    unsigned short* P = (unsigned short*)alloc((long long)NHEADS * S_LEN * S_LEN * 2);
    unsigned short* aout = (unsigned short*)alloc(nAct * 2); // [S,HIDDEN]

    // 1) fp32 -> bf16 conversions
    cvt_f32_bf16<<<dim3((unsigned)((nAct + 255) / 256)), dim3(256), 0, stream>>>(hs, hsb, nAct);
    cvt_f32_bf16<<<dim3((unsigned)((nW + 255) / 256)), dim3(256), 0, stream>>>(Wq, wqb, nW);
    cvt_f32_bf16<<<dim3((unsigned)((nW + 255) / 256)), dim3(256), 0, stream>>>(Wk, wkb, nW);
    cvt_f32_bf16<<<dim3((unsigned)((nW + 255) / 256)), dim3(256), 0, stream>>>(Wv, wvb, nW);
    cvt_f32_bf16<<<dim3((unsigned)((nW + 255) / 256)), dim3(256), 0, stream>>>(Wo, wob, nW);

    const dim3 gblk(256);   // 8 waves, 128x128 tile
    // 2) QKV projections: [S,HIDDEN] = hsb @ W^T  (grid: N/128 x M/128)
    gemm_bf16_wmma<<<dim3(HIDDEN_ / 128, S_LEN / 128, 1), gblk, 0, stream>>>(
        hsb, HIDDEN_, 0, wqb, HIDDEN_, 0, qf, HIDDEN_, 0,
        S_LEN, HIDDEN_, HIDDEN_, 1.0f, 0);
    gemm_bf16_wmma<<<dim3(HIDDEN_ / 128, S_LEN / 128, 1), gblk, 0, stream>>>(
        hsb, HIDDEN_, 0, wkb, HIDDEN_, 0, kf, HIDDEN_, 0,
        S_LEN, HIDDEN_, HIDDEN_, 1.0f, 0);
    gemm_bf16_wmma<<<dim3(HIDDEN_ / 128, S_LEN / 128, 1), gblk, 0, stream>>>(
        hsb, HIDDEN_, 0, wvb, HIDDEN_, 0, vf, HIDDEN_, 0,
        S_LEN, HIDDEN_, HIDDEN_, 1.0f, 0);

    // 3) RoPE + pack to [NH,S,HD] bf16; V -> [NH,HD,S] bf16
    rope_cvt<<<dim3(NHEADS * S_LEN * 64 / 256), dim3(256), 0, stream>>>(qf, kf, qh, kh);
    vT_cvt<<<dim3(NHEADS * HDIM * S_LEN / 256), dim3(256), 0, stream>>>(vf, vt);

    // 4) attn = (Q @ K^T) / sqrt(HD), causal fused (flags bit0); per-head z
    gemm_bf16_wmma<<<dim3(S_LEN / 128, S_LEN / 128, NHEADS), gblk, 0, stream>>>(
        qh, HDIM, (long long)S_LEN * HDIM,
        kh, HDIM, (long long)S_LEN * HDIM,
        attn, S_LEN, (long long)S_LEN * S_LEN,
        S_LEN, S_LEN, HDIM, 0.08838834764831845f, 1);

    // 5) sequential H2O scan, mask applied in place (one block per head)
    hh_scan<<<dim3(NHEADS), dim3(1024), 0, stream>>>(attn);

    // 6) final softmax -> bf16 probabilities
    softmax_rows<<<dim3(NHEADS * S_LEN), dim3(256), 0, stream>>>(attn, P);

    // 7) per-head PV: aout[s, h*HD + d] = P_h @ V_h  (bf16 out, flags bit1)
    gemm_bf16_wmma<<<dim3(HDIM / 128, S_LEN / 128, NHEADS), gblk, 0, stream>>>(
        P, S_LEN, (long long)S_LEN * S_LEN,
        vt, S_LEN, (long long)HDIM * S_LEN,
        aout, HIDDEN_, (long long)HDIM,
        S_LEN, HDIM, S_LEN, 1.0f, 2);

    // 8) output projection: d_out = aout @ Wo^T  (f32 out)
    gemm_bf16_wmma<<<dim3(HIDDEN_ / 128, S_LEN / 128, 1), gblk, 0, stream>>>(
        aout, HIDDEN_, 0, wob, HIDDEN_, 0, (float*)d_out, HIDDEN_, 0,
        S_LEN, HIDDEN_, HIDDEN_, 1.0f, 0);
}
